// EdgeMPNN_76768245449268
// MI455X (gfx1250) — matrix operand order repaired and checked by reading, not compiled, not run on hardware
//
#include <hip/hip_runtime.h>
#include <hip/hip_bf16.h>

// Problem constants from the reference
#define NN   512
#define VV   40
#define NF   32
#define EF   4
#define EMB  128
#define BIG_NEG (-1000000.0f)

typedef __attribute__((ext_vector_type(16))) _Float16 v16h;
typedef __attribute__((ext_vector_type(8)))  _Float16 v8h;
typedef __attribute__((ext_vector_type(8)))  float    v8f;

// ---------------------------------------------------------------------------
// Pack B = [W_att[128:256] | W_emb[128:256]] column-major f16: Bt[n][k], n<256,k<128
// ---------------------------------------------------------------------------
__global__ __launch_bounds__(128) void pack_B_kernel(const float* __restrict__ Wa,
                                                     const float* __restrict__ We,
                                                     _Float16* __restrict__ Bt) {
  int n = blockIdx.x;     // 0..255 (output column)
  int k = threadIdx.x;    // 0..127 (input/K dim)
  float v = (n < EMB) ? Wa[(EMB + k) * EMB + n] : We[(EMB + k) * EMB + (n - EMB)];
  Bt[n * EMB + k] = (_Float16)v;
}

// Wgs = Wg_att[0:128] + Wg_att[128:256]  (since cat3 = [gs, gs])
__global__ __launch_bounds__(128) void pack_Wg_kernel(const float* __restrict__ Wg_att,
                                                      float* __restrict__ Wgs) {
  int j = blockIdx.x, k = threadIdx.x;
  Wgs[j * EMB + k] = Wg_att[j * EMB + k] + Wg_att[(EMB + j) * EMB + k];
}

// slots[e*D + d] = in_eb + 1 at scattered positions (0 == empty == mask 0)
__global__ void scatter_slots_kernel(const int* __restrict__ igeb, const int* __restrict__ ige,
                                     const int* __restrict__ eb, int* __restrict__ slots,
                                     int NNZ, int D) {
  int i = blockIdx.x * 256 + threadIdx.x;
  if (i < NNZ) slots[igeb[i] * D + ige[i]] = eb[i] + 1;
}

// ---------------------------------------------------------------------------
// Per-edge pre-embedding: e_emb = tanh(e_in @ W_pre + b_pre)
// base_att = e_emb @ W_att[0:128] + b_att ; base_emb = e_emb @ W_emb[0:128] + b_emb
// ---------------------------------------------------------------------------
__global__ __launch_bounds__(128) void preembed_kernel(
    const float* __restrict__ nodes, const float* __restrict__ edg,
    const float* __restrict__ W_pre, const float* __restrict__ b_pre,
    const float* __restrict__ W_att, const float* __restrict__ b_att,
    const float* __restrict__ W_emb, const float* __restrict__ b_emb,
    const int* __restrict__ eb_b, const int* __restrict__ eb_n, const int* __restrict__ eb_nb,
    float* __restrict__ base_att, float* __restrict__ base_emb) {
  int e = blockIdx.x, k = threadIdx.x;
  __shared__ float sin_[2 * NF + EF];
  __shared__ float semb[EMB];
  int b = eb_b[e], n = eb_n[e], nb = eb_nb[e];
  if (k < NF)           sin_[k] = nodes[(b * VV + n) * NF + k];
  else if (k < 2 * NF)  sin_[k] = nodes[(b * VV + nb) * NF + (k - NF)];
  else if (k < 2 * NF + EF)
    sin_[k] = edg[(((size_t)b * VV + n) * VV + nb) * EF + (k - 2 * NF)];
  __syncthreads();
  float acc = b_pre[k];
#pragma unroll
  for (int i = 0; i < 2 * NF + EF; ++i) acc += sin_[i] * W_pre[i * EMB + k];
  semb[k] = tanhf(acc);
  __syncthreads();
  float a1 = b_att[k], a2 = b_emb[k];
  for (int j = 0; j < EMB; ++j) {
    float ej = semb[j];
    a1 += ej * W_att[j * EMB + k];
    a2 += ej * W_emb[j * EMB + k];
  }
  base_att[(size_t)e * EMB + k] = a1;
  base_emb[(size_t)e * EMB + k] = a2;
}

// ---------------------------------------------------------------------------
// QR[e, 0:256] = memH[e, 0:128] @ Bt^T   (WMMA f16 -> f32)
// Workgroup = 256 threads = 8 waves; 4 row-strips of 16 edges x 2 column halves.
// B (64 KB) is staged in LDS once per workgroup; A row-strip preloaded in regs.
// ---------------------------------------------------------------------------
__global__ __launch_bounds__(256) void gemm_mem_kernel(const _Float16* __restrict__ memH,
                                                       const _Float16* __restrict__ Bt,
                                                       float* __restrict__ QR, int E) {
  __shared__ _Float16 sB[256 * EMB];   // 64 KB

  const int tid  = threadIdx.x;
  const int lane = tid & 31;
  const int wave = tid >> 5;
  const int strip = wave >> 1;            // 0..3
  const int half  = wave & 1;             // 0..1 (which 128-col half)
  const int row0  = blockIdx.x * 64 + strip * 16;
  const int hi    = lane >> 4;            // 0 for lanes 0-15, 1 for 16-31
  const int rA    = min(row0 + (lane & 15), E - 1);   // clamped row for A loads

  // Cooperative copy of B into LDS: 4096 x 16B, 16 per thread
  {
    const int4* src = (const int4*)Bt;
    int4* dst = (int4*)sB;
#pragma unroll
    for (int i = 0; i < 16; ++i) dst[tid + i * 256] = src[tid + i * 256];
  }
  __syncthreads();

  // Preload entire A strip for this lane: elems 0..7 -> K = ks*32 + 8*hi + 0..7,
  // elems 8..15 -> +16  (ISA 7.12.2, 16-bit A 16x32). Whole row = 128 B / lane.
  v16h a[4];
#pragma unroll
  for (int ks = 0; ks < 4; ++ks) {
    const _Float16* pa = memH + (size_t)rA * EMB + ks * 32 + 8 * hi;
    v8h alo = *(const v8h*)(pa);
    v8h ahi = *(const v8h*)(pa + 16);
#pragma unroll
    for (int i = 0; i < 8; ++i) { a[ks][i] = alo[i]; a[ks][i + 8] = ahi[i]; }
  }

  v8f acc[8] = {};

#pragma unroll
  for (int ks = 0; ks < 4; ++ks) {
    // Load all 8 B tiles for this k-step from LDS, then issue 8 back-to-back WMMAs.
    // B tile: lane holds column n; elems 0..7 -> K = 8*hi + 0..7, elems 8..15 -> +16.
    v16h bt[8];
#pragma unroll
    for (int t = 0; t < 8; ++t) {
      const int n = (half * 8 + t) * 16 + (lane & 15);
      const _Float16* pb = sB + n * EMB + ks * 32 + 8 * hi;
      v8h blo = *(const v8h*)(pb);
      v8h bhi = *(const v8h*)(pb + 16);
#pragma unroll
      for (int i = 0; i < 8; ++i) { bt[t][i] = blo[i]; bt[t][i + 8] = bhi[i]; }
    }
#pragma unroll
    for (int t = 0; t < 8; ++t) {
      acc[t] = __builtin_amdgcn_wmma_f32_16x16x32_f16(false, a[ks], false, bt[t],
                                                      (short)0, acc[t], false, false);
    }
  }

  // C/D layout: VGPR r -> M = r + 8*hi, N = lane&15
#pragma unroll
  for (int t = 0; t < 8; ++t) {
    const int n = (half * 8 + t) * 16 + (lane & 15);
#pragma unroll
    for (int r = 0; r < 8; ++r) {
      int row = row0 + r + 8 * hi;
      if (row < E) QR[(size_t)row * 256 + n] = acc[t][r];
    }
  }
}

// ---------------------------------------------------------------------------
// Per-edge masked softmax over D slots + tanh message + weighted sum -> mem (f16)
// ---------------------------------------------------------------------------
__global__ __launch_bounds__(128) void combine_kernel(const float* __restrict__ base_att,
                                                      const float* __restrict__ base_emb,
                                                      const float* __restrict__ QR,
                                                      const int* __restrict__ slots,
                                                      _Float16* __restrict__ memH,
                                                      int E, int D) {
  int e = blockIdx.x, k = threadIdx.x;
  extern __shared__ float sm[];
  float* slog = sm;
  float* smsg = sm + (size_t)D * EMB;
  float ba = base_att[(size_t)e * EMB + k];
  float be = base_emb[(size_t)e * EMB + k];
  float mx = -3.0e38f;
  for (int d = 0; d < D; ++d) {
    int s = slots[e * D + d];
    float q = 0.f, r = 0.f;
    if (s > 0) {
      const float* p = QR + (size_t)(s - 1) * 256;
      q = p[k];
      r = p[EMB + k];
    }
    float l = ba + q + (s > 0 ? 0.f : BIG_NEG);
    slog[d * EMB + k] = l;
    smsg[d * EMB + k] = tanhf(be + r);
    mx = fmaxf(mx, l);
  }
  float Z = 0.f, acc = 0.f;
  for (int d = 0; d < D; ++d) {
    float w = __expf(slog[d * EMB + k] - mx);
    Z += w;
    acc += w * smsg[d * EMB + k];
  }
  memH[(size_t)e * EMB + k] = (_Float16)(acc / Z);
}

// graph_sets[b,n,:] += mem[e,:]
__global__ __launch_bounds__(128) void aggregate_kernel(const _Float16* __restrict__ memH,
                                                        const int* __restrict__ eb_b,
                                                        const int* __restrict__ eb_n,
                                                        float* __restrict__ gs, int E) {
  int e = blockIdx.x, k = threadIdx.x;
  int node = eb_b[e] * VV + eb_n[e];
  __hip_atomic_fetch_add(&gs[(size_t)node * EMB + k], (float)memH[(size_t)e * EMB + k],
                         __ATOMIC_RELAXED, __HIP_MEMORY_SCOPE_AGENT);
}

// ---------------------------------------------------------------------------
// Per-graph attention readout + final projection
// ---------------------------------------------------------------------------
__global__ __launch_bounds__(128) void readout_kernel(const float* __restrict__ gs,
                                                      const float* __restrict__ Wgs,
                                                      const float* __restrict__ bg_att,
                                                      const float* __restrict__ Wg_emb,
                                                      const float* __restrict__ bg_emb,
                                                      const float* __restrict__ W_out,
                                                      const float* __restrict__ b_out,
                                                      const float* __restrict__ node_mask,
                                                      float* __restrict__ out) {
  int b = blockIdx.x, k = threadIdx.x;
  __shared__ float sgs[VV * EMB];
  __shared__ float slog[VV * EMB];
  __shared__ float sge[EMB];
  for (int v = 0; v < VV; ++v) sgs[v * EMB + k] = gs[((size_t)b * VV + v) * EMB + k];
  __syncthreads();
  float mx = -3.0e38f;
  for (int v = 0; v < VV; ++v) {
    float acc = bg_att[k];
    for (int j = 0; j < EMB; ++j) acc += sgs[v * EMB + j] * Wgs[j * EMB + k];
    acc += (1.0f - node_mask[b * VV + v]) * BIG_NEG;
    slog[v * EMB + k] = acc;
    mx = fmaxf(mx, acc);
  }
  float Z = 0.f;
  for (int v = 0; v < VV; ++v) Z += __expf(slog[v * EMB + k] - mx);
  float ge = 0.f;
  for (int v = 0; v < VV; ++v) {
    float acc = bg_emb[k];
    for (int j = 0; j < EMB; ++j) acc += sgs[v * EMB + j] * Wg_emb[j * EMB + k];
    ge += __expf(slog[v * EMB + k] - mx) * tanhf(acc);
  }
  ge /= Z;
  sge[k] = ge;
  __syncthreads();
  float o = b_out[k];
  for (int j = 0; j < EMB; ++j) o += sge[j] * W_out[j * EMB + k];
  out[(size_t)b * EMB + k] = o;
}

// ---------------------------------------------------------------------------
extern "C" void kernel_launch(void* const* d_in, const int* in_sizes, int n_in,
                              void* d_out, int out_size, void* d_ws, size_t ws_size,
                              hipStream_t stream) {
  const float* nodes   = (const float*)d_in[0];
  const float* edg     = (const float*)d_in[1];
  const float* W_pre   = (const float*)d_in[2];
  const float* b_pre   = (const float*)d_in[3];
  const float* W_att   = (const float*)d_in[4];
  const float* b_att   = (const float*)d_in[5];
  const float* W_emb   = (const float*)d_in[6];
  const float* b_emb   = (const float*)d_in[7];
  const float* Wg_att  = (const float*)d_in[8];
  const float* bg_att  = (const float*)d_in[9];
  const float* Wg_emb  = (const float*)d_in[10];
  const float* bg_emb  = (const float*)d_in[11];
  const float* W_out   = (const float*)d_in[12];
  const float* b_out   = (const float*)d_in[13];
  // d_in[14] = ingoing_mask (E*D) -- reconstructed from slots instead
  const float* node_mask = (const float*)d_in[15];
  const int* eb_b   = (const int*)d_in[16];
  const int* eb_n   = (const int*)d_in[17];
  const int* eb_nb  = (const int*)d_in[18];
  const int* in_eb  = (const int*)d_in[19];
  const int* in_igeb= (const int*)d_in[20];
  const int* in_ige = (const int*)d_in[21];

  const int E   = in_sizes[16];
  const int D   = in_sizes[14] / E;
  const int NNZ = in_sizes[19];

  // workspace carve-out (256B aligned)
  char* p = (char*)d_ws;
  auto carve = [&](size_t bytes) -> void* {
    void* r = (void*)p;
    p += (bytes + 255) & ~(size_t)255;
    return r;
  };
  float*    base_att = (float*)   carve((size_t)E * EMB * 4);
  float*    base_emb = (float*)   carve((size_t)E * EMB * 4);
  float*    QR       = (float*)   carve((size_t)E * 256 * 4);
  _Float16* memH     = (_Float16*)carve((size_t)E * EMB * 2);
  int*      slots    = (int*)     carve((size_t)E * D * 4);
  _Float16* Bt       = (_Float16*)carve((size_t)256 * EMB * 2);
  float*    Wgs      = (float*)   carve((size_t)EMB * EMB * 4);
  float*    gs       = (float*)   carve((size_t)NN * VV * EMB * 4);

  hipMemsetAsync(slots, 0, (size_t)E * D * 4, stream);
  hipMemsetAsync(memH,  0, (size_t)E * EMB * 2, stream);
  hipMemsetAsync(gs,    0, (size_t)NN * VV * EMB * 4, stream);

  scatter_slots_kernel<<<(NNZ + 255) / 256, 256, 0, stream>>>(in_igeb, in_ige, in_eb,
                                                              slots, NNZ, D);
  pack_B_kernel<<<256, 128, 0, stream>>>(W_att, W_emb, Bt);
  pack_Wg_kernel<<<128, 128, 0, stream>>>(Wg_att, Wgs);
  preembed_kernel<<<E, 128, 0, stream>>>(nodes, edg, W_pre, b_pre, W_att, b_att,
                                         W_emb, b_emb, eb_b, eb_n, eb_nb,
                                         base_att, base_emb);
  const size_t shmem = (size_t)2 * D * EMB * 4;
  for (int s = 0; s < 3; ++s) {
    gemm_mem_kernel<<<(E + 63) / 64, 256, 0, stream>>>(memH, Bt, QR, E);
    combine_kernel<<<E, 128, shmem, stream>>>(base_att, base_emb, QR, slots, memH, E, D);
  }
  aggregate_kernel<<<E, 128, 0, stream>>>(memH, eb_b, eb_n, gs, E);
  readout_kernel<<<NN, 128, 0, stream>>>(gs, Wgs, bg_att, Wg_emb, bg_emb,
                                         W_out, b_out, node_mask, (float*)d_out);
}
